// GraphMLP_31069793419896
// MI455X (gfx1250) — compile-verified, matrix operand order checked by reading, not run on hardware
//
#include <hip/hip_runtime.h>
#include <math.h>

// ---------------------------------------------------------------------------
// GraphMLP GAT pipeline for MI455X (gfx1250, wave32, WMMA)
// B=16, N=2048, D_IN=256, E=32768 (+N self loops), H=2
// ---------------------------------------------------------------------------

#define NN    2048
#define BB    16
#define HH    2

typedef __attribute__((ext_vector_type(16))) _Float16 v16h;
typedef __attribute__((ext_vector_type(8)))  float    v8f;

__device__ __forceinline__ float leaky_f(float x) { return x > 0.f ? x : 0.2f * x; }
__device__ __forceinline__ float selu_f(float x) {
    const float sc = 1.0507009873554805f, al = 1.6732632423543772f;
    return x > 0.f ? sc * x : sc * al * (expf(x) - 1.f);
}

// ---------------------------------------------------------------------------
// Generic f32 GEMM via v_wmma_f32_16x16x32_f16 (f16 inputs, f32 accumulate).
// D[M,Nc] = A[M,K] @ B[K,Nc] (+ optional per-column bias). Row-major f32 in/out.
// One wave (32 threads) per 16x16 output tile. M%16==0, Nc%16==0, K%32==0.
// K and Nc compile-time -> fully unrolled wmma pipeline, all strided offsets
// fold into instruction IOFFSETs (no per-load 64-bit address chains).
// ---------------------------------------------------------------------------
template <int K, int Nc, bool HAS_BIAS>
__global__ void gemm_wmma_f16(const float* __restrict__ A,
                              const float* __restrict__ B,
                              float* __restrict__ D,
                              const float* __restrict__ bias,
                              int M)
{
    const int tileN = blockIdx.x;          // Nc / 16
    const int tileM = blockIdx.y;          // M  / 16
    const int lane  = threadIdx.x;         // 0..31 (wave32)
    const int row   = lane & 15;
    const int hi    = lane >> 4;

    const int    n  = tileN * 16 + row;
    const float  bv = HAS_BIAS ? bias[n] : 0.0f;
    const float* ar = A + (size_t)(tileM * 16 + row) * (size_t)K;
    const float* br = B + n;               // per-lane column base; rows via IOFFSET

    v8f acc = {};
#pragma unroll
    for (int k0 = 0; k0 < K; k0 += 32) {
        v16h af, bf;
#pragma unroll
        for (int i = 0; i < 8; ++i) {
            // 16-bit A 16x32 layout: vgpr i, lane-half hi -> K pair base
            const int k = ((i >= 4) ? 16 : 0) + hi * 8 + (i & 3) * 2;
            const float a0 = ar[k0 + k];
            const float a1 = ar[k0 + k + 1];
            af[2 * i]     = (_Float16)a0;
            af[2 * i + 1] = (_Float16)a1;
            // B 32x16 mirrored layout: col striped across lanes, rows constant-strided
            const float b0 = br[(k0 + k) * Nc];
            const float b1 = br[(k0 + k + 1) * Nc];
            bf[2 * i]     = (_Float16)b0;
            bf[2 * i + 1] = (_Float16)b1;
        }
        acc = __builtin_amdgcn_wmma_f32_16x16x32_f16(
            /*neg_a=*/false, af, /*neg_b=*/false, bf,
            /*c_mod=*/(short)0, acc, /*reuse_a=*/false, /*reuse_b=*/false);
    }
    float* dr = D + (size_t)(tileM * 16 + 8 * hi) * Nc + n;
#pragma unroll
    for (int r = 0; r < 8; ++r) {
        dr[r * Nc] = acc[r] + bv;
    }
}

// ---------------------------------------------------------------------------
// CSR build over destination nodes (edge structure shared by all layers/batches)
// ---------------------------------------------------------------------------
__global__ void zero_i32(int* __restrict__ p, int n)
{
    int i = blockIdx.x * blockDim.x + threadIdx.x;
    if (i < n) p[i] = 0;
}

__global__ void count_kernel(const int* __restrict__ dst, int E, int* __restrict__ counts)
{
    int e = blockIdx.x * blockDim.x + threadIdx.x;
    if (e < E) atomicAdd(&counts[dst[e]], 1);
}

__global__ void scan_kernel(const int* __restrict__ counts, int* __restrict__ rowptr, int n)
{
    if (threadIdx.x == 0 && blockIdx.x == 0) {
        int acc = 0;
        rowptr[0] = 0;
        for (int i = 0; i < n; ++i) { acc += counts[i]; rowptr[i + 1] = acc; }
    }
}

__global__ void fill_kernel(const int* __restrict__ src, const int* __restrict__ dst, int E,
                            const int* __restrict__ rowptr, int* __restrict__ fill,
                            int* __restrict__ col)
{
    int e = blockIdx.x * blockDim.x + threadIdx.x;
    if (e < E) {
        int d = dst[e];
        int pos = rowptr[d] + atomicAdd(&fill[d], 1);
        col[pos] = src[e];
    }
}

// ---------------------------------------------------------------------------
// Attention logits: es[r,h] = sum_c xw[r, h*C+c] * a_s[h*C+c] (ditto ed)
// ---------------------------------------------------------------------------
template <int C>
__global__ void att_dot_kernel(const float* __restrict__ xw,
                               const float* __restrict__ a_s,
                               const float* __restrict__ a_d,
                               float* __restrict__ es, float* __restrict__ ed, int rows)
{
    int i = blockIdx.x * blockDim.x + threadIdx.x;
    if (i >= rows * HH) return;
    int r = i >> 1, h = i & 1;
    const float* xr = xw + (size_t)r * (HH * C) + h * C;
    const float* as = a_s + h * C;
    const float* ad = a_d + h * C;
    float s = 0.f, d = 0.f;
#pragma unroll 4
    for (int c = 0; c < C; ++c) { s += xr[c] * as[c]; d += xr[c] * ad[c]; }
    es[i] = s;
    ed[i] = d;
}

// ---------------------------------------------------------------------------
// Layer 0 (decomposed): xw0[b,n] = xa[b] + xb[n]; softmax weights sum to 1 so
// out[b,d,h,:] = xa[b,h,:] + sum_e alpha_e * xb[src_e,h,:]
// Block = 128 threads = (h,c) with C=64; grid = (N, B). Self loop included.
// ---------------------------------------------------------------------------
__global__ void gat0_kernel(const float* __restrict__ xa,    // [16,128]
                            const float* __restrict__ xb,    // [2048,128]
                            const float* __restrict__ esa,   // [16,2]
                            const float* __restrict__ eda,   // [16,2]
                            const float* __restrict__ esb,   // [2048,2]
                            const float* __restrict__ edb,   // [2048,2]
                            const int* __restrict__ rowptr,
                            const int* __restrict__ col,
                            const float* __restrict__ bias,  // [64]
                            float* __restrict__ out)         // [B,N,64]
{
    const int d = blockIdx.x, b = blockIdx.y;
    const int t = threadIdx.x, h = t >> 6, c = t & 63;
    const int beg = rowptr[d], end = rowptr[d + 1];
    const float cst = esa[b * 2 + h] + eda[b * 2 + h] + edb[d * 2 + h];

    // pass 1: running max (self loop first)
    float mx = leaky_f(cst + esb[d * 2 + h]);
    for (int e = beg; e < end; ++e)
        mx = fmaxf(mx, leaky_f(cst + esb[col[e] * 2 + h]));

    // pass 2: exp-sum + weighted accumulation of xb
    float w   = expf(leaky_f(cst + esb[d * 2 + h]) - mx);
    float den = w;
    float acc = w * xb[d * 128 + t];
    for (int e = beg; e < end; ++e) {
        int s = col[e];
        float we = expf(leaky_f(cst + esb[s * 2 + h]) - mx);
        den += we;
        acc += we * xb[s * 128 + t];
    }
    acc = acc / den + xa[b * 128 + t];   // + xa since sum(alpha)==1

    __shared__ float sh[128];
    sh[t] = acc;
    __syncthreads();
    if (h == 0) {
        float v = 0.5f * (sh[c] + sh[64 + c]) + bias[c];
        out[((size_t)b * NN + d) * 64 + c] = selu_f(v);
    }
}

// ---------------------------------------------------------------------------
// Generic GAT layer (materialized xw[B,N,2C]); block = 2C threads, grid (N,B)
// ---------------------------------------------------------------------------
template <int C>
__global__ void gat_attn_kernel(const float* __restrict__ xw,
                                const float* __restrict__ es,   // [B,N,2]
                                const float* __restrict__ ed,   // [B,N,2]
                                const int* __restrict__ rowptr,
                                const int* __restrict__ col,
                                const float* __restrict__ bias, // [C]
                                float* __restrict__ out)        // [B,N,C]
{
    const int d = blockIdx.x, b = blockIdx.y;
    const int t = threadIdx.x, h = t / C, c = t % C;
    const size_t base = (size_t)b * NN;
    const float edv = ed[(base + d) * 2 + h];
    const int beg = rowptr[d], end = rowptr[d + 1];

    float mx = leaky_f(es[(base + d) * 2 + h] + edv);   // self loop
    for (int e = beg; e < end; ++e)
        mx = fmaxf(mx, leaky_f(es[(base + col[e]) * 2 + h] + edv));

    float w   = expf(leaky_f(es[(base + d) * 2 + h] + edv) - mx);
    float den = w;
    float acc = w * xw[(base + d) * (2 * C) + t];
    for (int e = beg; e < end; ++e) {
        int s = col[e];
        float we = expf(leaky_f(es[(base + s) * 2 + h] + edv) - mx);
        den += we;
        acc += we * xw[(base + s) * (2 * C) + t];
    }
    acc /= den;

    __shared__ float sh[2 * C];
    sh[t] = acc;
    __syncthreads();
    if (h == 0) {
        float v = 0.5f * (sh[c] + sh[C + c]) + bias[c];
        out[(base + d) * C + c] = selu_f(v);
    }
}

// ---------------------------------------------------------------------------
// Final projection: out[b,n] = h3[b,n,:16] . out_w + out_b
// ---------------------------------------------------------------------------
__global__ void out_kernel(const float* __restrict__ h3,
                           const float* __restrict__ ow,
                           const float* __restrict__ ob,
                           float* __restrict__ out, int total)
{
    int i = blockIdx.x * blockDim.x + threadIdx.x;
    if (i >= total) return;
    const float* r = h3 + (size_t)i * 16;
    float s = ob[0];
#pragma unroll
    for (int j = 0; j < 16; ++j) s += r[j] * ow[j];
    out[i] = s;
}

// ---------------------------------------------------------------------------
// Host-side launch
// ---------------------------------------------------------------------------
extern "C" void kernel_launch(void* const* d_in, const int* in_sizes, int n_in,
                              void* d_out, int out_size, void* d_ws, size_t ws_size,
                              hipStream_t stream)
{
    const float* x       = (const float*)d_in[0];   // [16,256]
    const float* node_x  = (const float*)d_in[1];   // [2048,128]
    const float* trans_w = (const float*)d_in[2];   // [128,64]
    const float* trans_b = (const float*)d_in[3];   // [64]
    const float* out_w   = (const float*)d_in[4];   // [16,1]
    const float* out_b   = (const float*)d_in[5];   // [1]
    const int*   eidx    = (const int*)d_in[6];     // [2,E]
    const float* g0_w    = (const float*)d_in[7];   // [320,2,64]
    const float* g0_as   = (const float*)d_in[8];
    const float* g0_ad   = (const float*)d_in[9];
    const float* g0_b    = (const float*)d_in[10];
    const float* g1_w    = (const float*)d_in[11];  // [64,2,32]
    const float* g1_as   = (const float*)d_in[12];
    const float* g1_ad   = (const float*)d_in[13];
    const float* g1_b    = (const float*)d_in[14];
    const float* g2_w    = (const float*)d_in[15];  // [32,2,16]
    const float* g2_as   = (const float*)d_in[16];
    const float* g2_ad   = (const float*)d_in[17];
    const float* g2_b    = (const float*)d_in[18];

    const int E = in_sizes[6] / 2;
    const int* src = eidx;
    const int* dst = eidx + E;

    // ---- workspace carve (floats first, then ints) ----
    float* p = (float*)d_ws;
    float* f_fused = p; p += (size_t)NN * 64;           // [2048,64]
    float* f_xa    = p; p += (size_t)BB * 128;          // [16,128]
    float* f_xb    = p; p += (size_t)NN * 128;          // [2048,128]
    float* f_esa   = p; p += (size_t)BB * HH;
    float* f_eda   = p; p += (size_t)BB * HH;
    float* f_esb   = p; p += (size_t)NN * HH;
    float* f_edb   = p; p += (size_t)NN * HH;
    float* f_bufA  = p; p += (size_t)BB * NN * 64;      // h1 / h2 / h3
    float* f_bufB  = p; p += (size_t)BB * NN * 64;      // xw1 / xw2
    float* f_es    = p; p += (size_t)BB * NN * HH;
    float* f_ed    = p; p += (size_t)BB * NN * HH;
    int* ip = (int*)p;
    int* i_counts = ip; ip += NN;
    int* i_rowptr = ip; ip += NN + 1;
    int* i_fill   = ip; ip += NN;
    int* i_col    = ip; ip += E;

    const int rows = BB * NN;                            // 32768

    // ---- CSR over dst ----
    zero_i32<<<(NN + 255) / 256, 256, 0, stream>>>(i_counts, NN);
    zero_i32<<<(NN + 255) / 256, 256, 0, stream>>>(i_fill, NN);
    count_kernel<<<(E + 255) / 256, 256, 0, stream>>>(dst, E, i_counts);
    scan_kernel<<<1, 1, 0, stream>>>(i_counts, i_rowptr, NN);
    fill_kernel<<<(E + 255) / 256, 256, 0, stream>>>(src, dst, E, i_rowptr, i_fill, i_col);

    // ---- fused = node_x @ trans_w + trans_b : [2048,128]x[128,64] ----
    gemm_wmma_f16<128, 64, true><<<dim3(64 / 16, NN / 16), 32, 0, stream>>>(
        node_x, trans_w, f_fused, trans_b, NN);

    // ---- layer-0 decomposition: xa = x @ W0[:256], xb = fused @ W0[256:] ----
    gemm_wmma_f16<256, 128, false><<<dim3(128 / 16, 1), 32, 0, stream>>>(
        x, g0_w, f_xa, nullptr, BB);
    gemm_wmma_f16<64, 128, false><<<dim3(128 / 16, NN / 16), 32, 0, stream>>>(
        f_fused, g0_w + (size_t)256 * 128, f_xb, nullptr, NN);

    // ---- layer-0 logit terms ----
    att_dot_kernel<64><<<(NN * HH + 255) / 256, 256, 0, stream>>>(
        f_xb, g0_as, g0_ad, f_esb, f_edb, NN);
    att_dot_kernel<64><<<1, 256, 0, stream>>>(
        f_xa, g0_as, g0_ad, f_esa, f_eda, BB);

    // ---- layer-0 attention + aggregation -> h1 [B,N,64] ----
    gat0_kernel<<<dim3(NN, BB), 128, 0, stream>>>(
        f_xa, f_xb, f_esa, f_eda, f_esb, f_edb, i_rowptr, i_col, g0_b, f_bufA);

    // ---- layer 1: xw1 = h1 @ W1 : [32768,64]x[64,64] ----
    gemm_wmma_f16<64, 64, false><<<dim3(64 / 16, rows / 16), 32, 0, stream>>>(
        f_bufA, g1_w, f_bufB, nullptr, rows);
    att_dot_kernel<32><<<(rows * HH + 255) / 256, 256, 0, stream>>>(
        f_bufB, g1_as, g1_ad, f_es, f_ed, rows);
    gat_attn_kernel<32><<<dim3(NN, BB), 64, 0, stream>>>(
        f_bufB, f_es, f_ed, i_rowptr, i_col, g1_b, f_bufA);   // h2 -> bufA

    // ---- layer 2: xw2 = h2 @ W2 : [32768,32]x[32,32] ----
    gemm_wmma_f16<32, 32, false><<<dim3(32 / 16, rows / 16), 32, 0, stream>>>(
        f_bufA, g2_w, f_bufB, nullptr, rows);
    att_dot_kernel<16><<<(rows * HH + 255) / 256, 256, 0, stream>>>(
        f_bufB, g2_as, g2_ad, f_es, f_ed, rows);
    gat_attn_kernel<16><<<dim3(NN, BB), 32, 0, stream>>>(
        f_bufB, f_es, f_ed, i_rowptr, i_col, g2_b, f_bufA);   // h3 -> bufA

    // ---- output projection ----
    out_kernel<<<(rows + 255) / 256, 256, 0, stream>>>(
        f_bufA, out_w, out_b, (float*)d_out, rows);
}
